// SentenceTagger_66571993088723
// MI455X (gfx1250) — matrix operand order repaired and checked by reading, not compiled, not run on hardware
//
#include <hip/hip_runtime.h>
#include <hip/hip_bf16.h>

#define Bn   64
#define Sn   256
#define LCn  20
#define DWn  300
#define DCn  30
#define NFn  50
#define KWn  3
#define DINn 350
#define DKp  352           // DIN padded to 11*32 for WMMA K-steps
#define Hn   256
#define G4   1024          // 4*H
#define Tn   12
#define ROWS (Bn * Sn)     // 16384, row index r = s*B + b

typedef __attribute__((ext_vector_type(16))) _Float16 v16h;
typedef __attribute__((ext_vector_type(8)))  _Float16 v8h;
typedef __attribute__((ext_vector_type(8)))  float    v8f;

// ---------------- WMMA fragment helpers (16x16x32 f16; contiguous k order, same in A and B) ----------------
__device__ __forceinline__ v16h make_frag(v8h a0, v8h a1) {
  v16h r;
#pragma unroll
  for (int i = 0; i < 8; ++i) { r[i] = a0[i]; r[i + 8] = a1[i]; }
  return r;
}
// A-matrix 16x32 f16: lane (hi = lane>>4) holds K in [hi*16, hi*16+16) contiguous
__device__ __forceinline__ v16h frag_a(const _Float16* rowk, int hi) {
  v8h a0 = *(const v8h*)(rowk + hi * 16);
  v8h a1 = *(const v8h*)(rowk + hi * 16 + 8);
  return make_frag(a0, a1);
}
// B-matrix 32x16 f16: lane holds 16 contiguous K of its column (n-major weight storage)
__device__ __forceinline__ v16h frag_b(const _Float16* p) {
  v8h b0 = *(const v8h*)(p);
  v8h b1 = *(const v8h*)(p + 8);
  return make_frag(b0, b1);
}
__device__ __forceinline__ float sigm(float x) { return 1.0f / (1.0f + __expf(-x)); }

// ---------------- K1: weight conversion fp32 -> fp16 (padded), bias fuse ----------------
__global__ void prep_kernel(const float* __restrict__ wihF, const float* __restrict__ wihB,
                            const float* __restrict__ whhF, const float* __restrict__ whhB,
                            const float* __restrict__ bihF, const float* __restrict__ bhhF,
                            const float* __restrict__ bihB, const float* __restrict__ bhhB,
                            _Float16* __restrict__ wihFh, _Float16* __restrict__ wihBh,
                            _Float16* __restrict__ whhFh, _Float16* __restrict__ whhBh,
                            float* __restrict__ biasF, float* __restrict__ biasB,
                            float* __restrict__ out0) {
  int idx = blockIdx.x * blockDim.x + threadIdx.x;
  if (idx < G4 * DKp) {
    int n = idx / DKp, k = idx % DKp;
    wihFh[idx] = (_Float16)((k < DINn) ? wihF[n * DINn + k] : 0.f);
    wihBh[idx] = (_Float16)((k < DINn) ? wihB[n * DINn + k] : 0.f);
  }
  if (idx < G4 * Hn) {
    whhFh[idx] = (_Float16)whhF[idx];
    whhBh[idx] = (_Float16)whhB[idx];
  }
  if (idx < G4) {
    biasF[idx] = bihF[idx] + bhhF[idx];
    biasB[idx] = bihB[idx] + bhhB[idx];
  }
  if (idx == 0) out0[0] = 0.f;
}

// ---------------- K2: embedding gather + char-CNN maxpool -> emb_h[ROWS][352] f16 ------
__global__ __launch_bounds__(256) void embed_kernel(
    const int* __restrict__ words, const int* __restrict__ chars,
    const float* __restrict__ glove, const float* __restrict__ cemb,
    const float* __restrict__ convw, const float* __restrict__ convb,
    _Float16* __restrict__ embh) {
  __shared__ float cw[NFn * DCn * KWn];   // 4500 floats, shared by 4 rows
  __shared__ float cb[NFn];
  __shared__ float ce[4][LCn * DCn];      // per-sub char embeddings
  const int tid = threadIdx.x;
  const int sub = tid >> 6, tl = tid & 63;
  const int row = blockIdx.x * 4 + sub;   // r = s*B + b
  const int s = row / Bn, b = row % Bn;

  for (int i = tid; i < NFn * DCn * KWn; i += 256) cw[i] = convw[i];
  for (int i = tid; i < NFn; i += 256) cb[i] = convb[i];
  for (int i = tl; i < LCn * DCn; i += 64) {
    int lc = i / DCn, dc = i % DCn;
    int cid = chars[((size_t)(b * Sn + s)) * LCn + lc];
    ce[sub][i] = cemb[cid * DCn + dc];
  }
  __syncthreads();

  _Float16* erow = embh + (size_t)row * DKp;
  const float* grow = glove + (size_t)words[b * Sn + s] * DWn;
  for (int d = tl; d < DKp; d += 64) {
    if (d < DWn)        erow[d] = (_Float16)grow[d];
    else if (d >= DINn) erow[d] = (_Float16)0.0f;   // pad K 350..351
  }
  if (tl < NFn) {
    const float* wf = cw + tl * (DCn * KWn);
    float mx = -1e30f;
    for (int pos = 0; pos < LCn - KWn + 1; ++pos) {
      float a = cb[tl];
#pragma unroll
      for (int k = 0; k < KWn; ++k) {
        const float* cp = ce[sub] + (pos + k) * DCn;
#pragma unroll
        for (int c = 0; c < DCn; ++c) a += cp[c] * wf[c * KWn + k];
      }
      mx = fmaxf(mx, a);
    }
    erow[DWn + tl] = (_Float16)mx;
  }
}

// ---------------- K3: xw = emb @ w_ih^T (both directions), big WMMA GEMM --------------
// Each wave: one 16-row M tile x four 16-col N tiles; K = 352 -> 11 WMMA k-steps.
// Output stored TRANSPOSED as xwT[s][n(4H)][b(64)] so the recurrence kernel reads its
// per-gate slab with single global_load_b128 per lane (8 contiguous batch rows).
__global__ __launch_bounds__(256) void gemm_xw_kernel(
    const _Float16* __restrict__ emb,
    const _Float16* __restrict__ wihF, const _Float16* __restrict__ wihB,
    _Float16* __restrict__ xwF, _Float16* __restrict__ xwB) {
  const int dir = blockIdx.y;
  const _Float16* wih = dir ? wihB : wihF;
  _Float16* xw = dir ? xwB : xwF;
  const int wid  = blockIdx.x * 8 + (threadIdx.x >> 5);
  const int lane = threadIdx.x & 31;
  const int hi = lane >> 4, lo = lane & 15;
  const int mt = wid >> 4;        // 0..1023
  const int n4 = wid & 15;        // 0..15 -> 4 N tiles each

  const v8f z = {0.f, 0.f, 0.f, 0.f, 0.f, 0.f, 0.f, 0.f};
  v8f acc[4] = {z, z, z, z};
  const _Float16* arow = emb + (size_t)(mt * 16 + lo) * DKp;
#pragma unroll
  for (int kt = 0; kt < 11; ++kt) {
    v16h a = frag_a(arow + kt * 32, hi);
#pragma unroll
    for (int q = 0; q < 4; ++q) {
      int n = (n4 * 4 + q) * 16 + lo;
      v16h bf = frag_b(wih + (size_t)n * DKp + kt * 32 + hi * 16);
      acc[q] = __builtin_amdgcn_wmma_f32_16x16x32_f16(false, a, false, bf,
                                                      (short)0, acc[q], false, false);
    }
  }
  // C-tile rows m = mt*16 + hi*8 + r (r=0..7 contiguous, within one 64-row s block)
  const int m0 = mt * 16 + hi * 8;
  const int sIdx = m0 >> 6, b0 = m0 & 63;
#pragma unroll
  for (int q = 0; q < 4; ++q) {
    int n = (n4 * 4 + q) * 16 + lo;
    v8h pk;
#pragma unroll
    for (int r = 0; r < 8; ++r) pk[r] = (_Float16)acc[q][r];
    *(v8h*)(xw + ((size_t)(sIdx * G4 + n)) * Bn + b0) = pk;   // one b128 store
  }
}

// ---------------- K4: persistent BiLSTM recurrence ------------------------------------
// Recurrence is serial in s but independent per batch row -> split batch 4 ways:
// 8 workgroups = 4 batch-groups x 2 directions, each on its own WGP.
// WG: 512 threads = 16 waves; wave jt owns j-block jt*16..+16 and all 4 gates for the
// WG's 16 batch rows, so the LSTM cell update is fully in registers (c-state in VGPRs,
// h-state in 8KB LDS). w_hh (1MB f16 both dirs) streams from the 192MB L2 each step.
__global__ __launch_bounds__(512) void lstm_kernel(
    const _Float16* __restrict__ xwF, const _Float16* __restrict__ xwB,
    const _Float16* __restrict__ whhFh, const _Float16* __restrict__ whhBh,
    const float* __restrict__ biasF, const float* __restrict__ biasB,
    _Float16* __restrict__ hF, _Float16* __restrict__ hB) {
  const int dir = blockIdx.x >> 2;        // 0 fwd, 1 bwd
  const int bg  = blockIdx.x & 3;         // batch group: rows bg*16..bg*16+15
  const _Float16* xw   = dir ? xwB   : xwF;    // [S][4H][64] transposed
  const _Float16* whh  = dir ? whhBh : whhFh;
  const float*    bias = dir ? biasB : biasF;
  _Float16*       hout = dir ? hB    : hF;

  __shared__ _Float16 hsh[16 * Hn];       // 8 KB: this WG's h rows, [m_loc][j] f16
  const int tid = threadIdx.x;
  for (int i = tid; i < 16 * Hn; i += 512) hsh[i] = (_Float16)0.0f;

  const int wave = tid >> 5, lane = tid & 31;
  const int hi = lane >> 4, lo = lane & 15;
  const int jt = wave;                    // 0..15 -> j block
  const int j  = jt * 16 + lo;

  float bi4[4];
#pragma unroll
  for (int g = 0; g < 4; ++g) bi4[g] = bias[g * Hn + j];

  const v8f z = {0.f, 0.f, 0.f, 0.f, 0.f, 0.f, 0.f, 0.f};
  v8f cst = z;                            // cell state (8 (m,j) cells per lane)
  __syncthreads();

  for (int step = 0; step < Sn; ++step) {
    const int s = dir ? (Sn - 1 - step) : step;
    v8f acc[4] = {z, z, z, z};
#pragma unroll
    for (int kt = 0; kt < 8; ++kt) {
      v16h a = frag_a(hsh + (size_t)lo * Hn + kt * 32, hi);   // ds_read_b128 x2
#pragma unroll
      for (int g = 0; g < 4; ++g) {
        int n = g * Hn + j;
        v16h bf = frag_b(whh + (size_t)n * Hn + kt * 32 + hi * 16);
        acc[g] = __builtin_amdgcn_wmma_f32_16x16x32_f16(false, a, false, bf,
                                                        (short)0, acc[g], false, false);
      }
    }
    // xw slab for this lane: 8 contiguous batch rows per gate -> 4x global_load_b128
    const int bBase = bg * 16 + hi * 8;
    v8h xv[4];
#pragma unroll
    for (int g = 0; g < 4; ++g)
      xv[g] = *(const v8h*)(xw + ((size_t)(s * G4 + g * Hn + j)) * Bn + bBase);
    if (step + 1 < Sn) {                  // prefetch next step's slab (global_prefetch_b8)
      const int s2 = dir ? (s - 1) : (s + 1);
      __builtin_prefetch(xw + ((size_t)(s2 * G4 + j)) * Bn + bBase, 0, 0);
    }
    __syncthreads();   // all waves done reading h(t-1) from LDS
#pragma unroll
    for (int r = 0; r < 8; ++r) {
      int mloc = hi * 8 + r;              // local batch row 0..15
      float gi = acc[0][r] + (float)xv[0][r] + bi4[0];
      float gf = acc[1][r] + (float)xv[1][r] + bi4[1];
      float gg = acc[2][r] + (float)xv[2][r] + bi4[2];
      float go = acc[3][r] + (float)xv[3][r] + bi4[3];
      float iv = sigm(gi), fv = sigm(gf), ov = sigm(go);
      float gv = tanhf(gg);
      float c = fv * cst[r] + iv * gv;
      cst[r] = c;
      float h = ov * tanhf(c);
      hsh[mloc * Hn + j] = (_Float16)h;
      hout[((size_t)(s * Bn + bg * 16 + mloc)) * Hn + j] = (_Float16)h;
    }
    __syncthreads();   // h(t) visible before next step's fragment reads
  }
}

// ---------------- K5: emissions = [hf, hb] @ fc_w^T + fc_b ----------------------------
__global__ void fc_kernel(const _Float16* __restrict__ hF, const _Float16* __restrict__ hB,
                          const float* __restrict__ fcw, const float* __restrict__ fcb,
                          float* __restrict__ em) {
  int id = blockIdx.x * blockDim.x + threadIdx.x;
  if (id >= ROWS * Tn) return;
  int row = id / Tn, t = id % Tn;
  const _Float16* hf = hF + (size_t)row * Hn;
  const _Float16* hb = hB + (size_t)row * Hn;
  const float* w = fcw + t * (2 * Hn);
  float s = fcb[t];
#pragma unroll 4
  for (int k = 0; k < Hn; ++k) s += (float)hf[k] * w[k] + (float)hb[k] * w[Hn + k];
  em[id] = s;
}

// ---------------- K6: CRF forward (NLL) + Viterbi, one wave per batch element ---------
__global__ void crf_kernel(const float* __restrict__ em, const int* __restrict__ tags,
                           const float* __restrict__ startt, const float* __restrict__ endt,
                           const float* __restrict__ trans, unsigned char* __restrict__ bp,
                           float* __restrict__ lossPart, float* __restrict__ out) {
  const int b = blockIdx.x;
  const int t = threadIdx.x;
  __shared__ float tr[Tn * Tn];
  __shared__ float alpha[Tn], nxt[Tn], va[Tn], vn[Tn];
  for (int i = t; i < Tn * Tn; i += 32) tr[i] = trans[i];
  __syncthreads();
  if (t < Tn) {
    float e0 = em[(size_t)(0 * Bn + b) * Tn + t];
    alpha[t] = startt[t] + e0;
    va[t]    = startt[t] + e0;
  }
  __syncthreads();
  for (int s = 1; s < Sn; ++s) {
    if (t < Tn) {
      float e = em[(size_t)(s * Bn + b) * Tn + t];
      float mx = -1e30f;
#pragma unroll
      for (int i = 0; i < Tn; ++i) mx = fmaxf(mx, alpha[i] + tr[i * Tn + t]);
      float sum = 0.f;
#pragma unroll
      for (int i = 0; i < Tn; ++i) sum += __expf(alpha[i] + tr[i * Tn + t] - mx);
      nxt[t] = mx + __logf(sum) + e;
      float best = -1e30f; int bi = 0;
#pragma unroll
      for (int i = 0; i < Tn; ++i) {
        float v = va[i] + tr[i * Tn + t];
        if (v > best) { best = v; bi = i; }
      }
      vn[t] = best + e;
      bp[((size_t)b * Sn + s) * Tn + t] = (unsigned char)bi;
    }
    __syncthreads();
    if (t < Tn) { alpha[t] = nxt[t]; va[t] = vn[t]; }
    __syncthreads();
  }
  if (t == 0) {
    float mx = -1e30f;
    for (int i = 0; i < Tn; ++i) mx = fmaxf(mx, alpha[i] + endt[i]);
    float sum = 0.f;
    for (int i = 0; i < Tn; ++i) sum += __expf(alpha[i] + endt[i] - mx);
    float logZ = mx + __logf(sum);
    // gold path score
    int prev = tags[b * Sn + 0];
    float num = startt[prev];
    for (int s = 0; s < Sn; ++s) {
      int tg = tags[b * Sn + s];
      num += em[(size_t)(s * Bn + b) * Tn + tg];
      if (s > 0) num += tr[prev * Tn + tg];
      prev = tg;
    }
    num += endt[prev];
    lossPart[b] = logZ - num;
    // Viterbi backtrack
    float bbest = -1e30f; int cur = 0;
    for (int i = 0; i < Tn; ++i) {
      float v = va[i] + endt[i];
      if (v > bbest) { bbest = v; cur = i; }
    }
    out[1 + b * Sn + (Sn - 1)] = (float)cur;
    for (int s = Sn - 1; s >= 1; --s) {
      cur = bp[((size_t)b * Sn + s) * Tn + cur];
      out[1 + b * Sn + (s - 1)] = (float)cur;
    }
  }
}

// ---------------- K7: deterministic loss reduction ------------------------------------
__global__ void loss_sum_kernel(const float* __restrict__ part, float* __restrict__ out) {
  if (threadIdx.x == 0 && blockIdx.x == 0) {
    float s = 0.f;
    for (int i = 0; i < Bn; ++i) s += part[i];
    out[0] = s;
  }
}

// ---------------- host launcher -------------------------------------------------------
extern "C" void kernel_launch(void* const* d_in, const int* in_sizes, int n_in,
                              void* d_out, int out_size, void* d_ws, size_t ws_size,
                              hipStream_t stream) {
  (void)in_sizes; (void)n_in; (void)out_size; (void)ws_size;
  const int*   words  = (const int*)d_in[0];
  const int*   chars  = (const int*)d_in[1];
  const int*   tags   = (const int*)d_in[2];
  const float* glove  = (const float*)d_in[3];
  const float* cemb   = (const float*)d_in[4];
  const float* convw  = (const float*)d_in[5];
  const float* convb  = (const float*)d_in[6];
  const float* wihF   = (const float*)d_in[7];
  const float* whhF   = (const float*)d_in[8];
  const float* bihF   = (const float*)d_in[9];
  const float* bhhF   = (const float*)d_in[10];
  const float* wihB   = (const float*)d_in[11];
  const float* whhB   = (const float*)d_in[12];
  const float* bihB   = (const float*)d_in[13];
  const float* bhhB   = (const float*)d_in[14];
  const float* fcw    = (const float*)d_in[15];
  const float* fcb    = (const float*)d_in[16];
  const float* startt = (const float*)d_in[17];
  const float* endt   = (const float*)d_in[18];
  const float* trans  = (const float*)d_in[19];
  float* out = (float*)d_out;

  // workspace carve-up (256B aligned)
  size_t off = 0;
  auto carve = [&](size_t bytes) -> char* {
    char* p = (char*)d_ws + off;
    off = (off + bytes + 255) & ~(size_t)255;
    return p;
  };
  _Float16* embh  = (_Float16*)carve((size_t)ROWS * DKp * 2);   // 11.5 MB
  _Float16* wihFh = (_Float16*)carve((size_t)G4 * DKp * 2);
  _Float16* wihBh = (_Float16*)carve((size_t)G4 * DKp * 2);
  _Float16* whhFh = (_Float16*)carve((size_t)G4 * Hn * 2);
  _Float16* whhBh = (_Float16*)carve((size_t)G4 * Hn * 2);
  float*    biasF = (float*)carve((size_t)G4 * 4);
  float*    biasB = (float*)carve((size_t)G4 * 4);
  _Float16* xwF   = (_Float16*)carve((size_t)ROWS * G4 * 2);    // 32 MB, [S][4H][B]
  _Float16* xwB   = (_Float16*)carve((size_t)ROWS * G4 * 2);    // 32 MB, [S][4H][B]
  _Float16* hF    = (_Float16*)carve((size_t)ROWS * Hn * 2);    // 8 MB
  _Float16* hB    = (_Float16*)carve((size_t)ROWS * Hn * 2);    // 8 MB
  float*    emis  = (float*)carve((size_t)ROWS * Tn * 4);       // 0.8 MB
  unsigned char* bp = (unsigned char*)carve((size_t)Bn * Sn * Tn);
  float*    lossPart = (float*)carve((size_t)Bn * 4);

  prep_kernel<<<(G4 * DKp + 255) / 256, 256, 0, stream>>>(
      wihF, wihB, whhF, whhB, bihF, bhhF, bihB, bhhB,
      wihFh, wihBh, whhFh, whhBh, biasF, biasB, out);

  embed_kernel<<<ROWS / 4, 256, 0, stream>>>(words, chars, glove, cemb, convw, convb, embh);

  dim3 gGemm(2048, 2);   // 16384 waves/dir: 1024 M-tiles x 16 wave-columns
  gemm_xw_kernel<<<gGemm, 256, 0, stream>>>(embh, wihFh, wihBh, xwF, xwB);

  lstm_kernel<<<8, 512, 0, stream>>>(xwF, xwB, whhFh, whhBh, biasF, biasB, hF, hB);

  fc_kernel<<<(ROWS * Tn + 255) / 256, 256, 0, stream>>>(hF, hB, fcw, fcb, emis);

  crf_kernel<<<Bn, 32, 0, stream>>>(emis, tags, startt, endt, trans, bp, lossPart, out);

  loss_sum_kernel<<<1, 64, 0, stream>>>(lossPart, out);
}
